// CapsuleLinear_18056042512757
// MI455X (gfx1250) — matrix operand order repaired and checked by reading, not compiled, not run on hardware
//
#include <hip/hip_runtime.h>

#define B_ 32
#define I_ 512
#define L_ 64
#define O_ 128
#define J_ 64
#define STRIDE 68            // floats per priors row in LDS (16B aligned, low conflicts)
#define EPS_ 1e-12f

typedef __attribute__((ext_vector_type(16))) __bf16         v16bf;
typedef __attribute__((ext_vector_type(16))) unsigned short v16us;
typedef __attribute__((ext_vector_type(8)))  unsigned short v8us;
typedef __attribute__((ext_vector_type(8)))  float          v8f;

__device__ __forceinline__ unsigned short f2bf(float f) {
    unsigned u = __builtin_bit_cast(unsigned, f);
    u += 0x7FFFu + ((u >> 16) & 1u);      // round-to-nearest-even
    return (unsigned short)(u >> 16);
}

// ---- Prepass: f32 -> bf16 bulk convert (runs once per launch; L2/HBM bound, ~1us) ----
__global__ __launch_bounds__(256)
void cvt_bf16_kernel(const float* __restrict__ src, unsigned short* __restrict__ dst, int n4) {
    int idx = blockIdx.x * blockDim.x + threadIdx.x;
    if (idx < n4) {
        float4 f = ((const float4*)src)[idx];
        ushort4 r;
        r.x = f2bf(f.x); r.y = f2bf(f.y); r.z = f2bf(f.z); r.w = f2bf(f.w);
        ((ushort4*)dst)[idx] = r;
    }
}

// LDS floats: priors 512*68 | out 64 | out_n 64 | red 1024 | wred 8 | coef 512
#define LDS_FLOATS (I_ * STRIDE + 64 + 64 + 1024 + 8 + 512)

template <bool PRECONV>
__global__ __launch_bounds__(256)
void capsule_kmeans_kernel(const float* __restrict__ x,            // (B, I, L) f32
                           const float* __restrict__ w,            // (O, J, L) f32
                           const unsigned short* __restrict__ xb16,// (B, I, L) bf16 bits
                           const unsigned short* __restrict__ wb16,// (O, J, L) bf16 bits
                           float* __restrict__ out)                // (B, O, J)
{
    extern __shared__ float lds[];
    float* priors = lds;                      // 512 x 68
    float* outv   = lds + I_ * STRIDE;        // 64
    float* outn   = outv + 64;                // 64
    float* red    = outn + 64;                // 16 x 64
    float* wred   = red + 1024;               // 8
    float* coef   = wred + 8;                 // 512

    const int tid  = threadIdx.x;
    const int lane = tid & 31;
    const int wv   = tid >> 5;                // wave id 0..7
    const int o    = blockIdx.x;
    const int b    = blockIdx.y;

    // ============ Phase 1: priors(512x64) = X(512x64) * W_o^T via bf16 WMMA ============
    const int r = lane & 15;                  // row (A) / col (B) within 16-tile
    const int h = lane >> 4;                  // lane half

    v8f acc[4][4];
    #pragma unroll
    for (int mt = 0; mt < 4; ++mt)
        #pragma unroll
        for (int nt = 0; nt < 4; ++nt)
            acc[mt][nt] = v8f{0.f,0.f,0.f,0.f,0.f,0.f,0.f,0.f};

    #pragma unroll
    for (int ks = 0; ks < 2; ++ks) {
        v16bf afrag[4], bfrag[4];
        // A: contiguous layout -> lane holds K {kb..kb+15}, kb = 32*ks + 16*h (matches B)
        const int kbA = ks * 32 + h * 16;
        #pragma unroll
        for (int mt = 0; mt < 4; ++mt) {
            const size_t rowoff = (size_t)((b * I_) + wv * 64 + mt * 16 + r) * L_ + kbA;
            if constexpr (PRECONV) {
                v8us lo = *(const v8us*)(xb16 + rowoff);
                v8us hi = *(const v8us*)(xb16 + rowoff + 8);
                v16us t = __builtin_shufflevector(lo, hi, 0,1,2,3,4,5,6,7,8,9,10,11,12,13,14,15);
                afrag[mt] = __builtin_bit_cast(v16bf, t);
            } else {
                const float* p = x + rowoff;
                v16us t;
                #pragma unroll
                for (int e = 0; e < 8; ++e) { t[e] = f2bf(p[e]); t[e + 8] = f2bf(p[8 + e]); }
                afrag[mt] = __builtin_bit_cast(v16bf, t);
            }
        }
        // B: 32x16 layout -> lanes 0-15: K 0..15, lanes 16-31: K 16..31 (contiguous)
        const int kbB = ks * 32 + h * 16;
        #pragma unroll
        for (int nt = 0; nt < 4; ++nt) {
            const size_t rowoff = (size_t)((o * J_) + nt * 16 + r) * L_ + kbB;
            if constexpr (PRECONV) {
                v16us t = *(const v16us*)(wb16 + rowoff);
                bfrag[nt] = __builtin_bit_cast(v16bf, t);
            } else {
                const float* p = w + rowoff;
                v16us t;
                #pragma unroll
                for (int e = 0; e < 16; ++e) t[e] = f2bf(p[e]);
                bfrag[nt] = __builtin_bit_cast(v16bf, t);
            }
        }
        #pragma unroll
        for (int mt = 0; mt < 4; ++mt)
            #pragma unroll
            for (int nt = 0; nt < 4; ++nt)
                acc[mt][nt] = __builtin_amdgcn_wmma_f32_16x16x32_bf16(
                    false, afrag[mt], false, bfrag[nt],
                    (short)0, acc[mt][nt], false, false);
    }

    // Store accumulators to LDS priors.  C layout: VGPR rr -> M = rr + 8*h, N = lane&15
    #pragma unroll
    for (int mt = 0; mt < 4; ++mt)
        #pragma unroll
        for (int nt = 0; nt < 4; ++nt)
            #pragma unroll
            for (int rr = 0; rr < 8; ++rr) {
                int row = wv * 64 + mt * 16 + rr + 8 * h;
                int col = nt * 16 + r;
                priors[row * STRIDE + col] = acc[mt][nt][rr];
            }
    __syncthreads();

    // ============ Phase 2: k-means routing entirely in LDS ============
    const int jg = tid & 15;     // column group: 4 cols at 4*jg
    const int qq = tid >> 4;     // row group: 32 rows at 32*qq

    // out_j = mean over i (float4 column partial sums)
    {
        float4 s = {0.f, 0.f, 0.f, 0.f};
        for (int i = qq * 32; i < qq * 32 + 32; ++i) {
            float4 p = *(const float4*)(priors + i * STRIDE + jg * 4);
            s.x += p.x; s.y += p.y; s.z += p.z; s.w += p.w;
        }
        *(float4*)(red + qq * 64 + jg * 4) = s;
    }
    __syncthreads();
    if (tid < 64) {
        float s = 0.f;
        #pragma unroll
        for (int q2 = 0; q2 < 16; ++q2) s += red[q2 * 64 + tid];
        outv[tid] = s * (1.f / 512.f);
    }
    __syncthreads();

    for (int it = 0; it < 3; ++it) {
        // L2-normalize out over J (wave 0 handles it)
        if (tid < 32) {
            float a0 = outv[tid], a1 = outv[tid + 32];
            float ss = a0 * a0 + a1 * a1;
            #pragma unroll
            for (int m = 16; m; m >>= 1) ss += __shfl_xor(ss, m, 32);
            float inv = 1.f / fmaxf(sqrtf(ss), EPS_);
            outn[tid]      = a0 * inv;
            outn[tid + 32] = a1 * inv;
        }
        __syncthreads();

        // logits_i = <priors[i,:], out_n> ; each thread does rows tid and tid+256
        float l0 = 0.f, l1 = 0.f;
        {
            const float4* row0 = (const float4*)(priors + tid * STRIDE);
            const float4* row1 = (const float4*)(priors + (tid + 256) * STRIDE);
            const float4* on4  = (const float4*)outn;
            #pragma unroll
            for (int jj = 0; jj < 16; ++jj) {
                float4 o4 = on4[jj];
                float4 p0 = row0[jj];
                float4 p1 = row1[jj];
                l0 += p0.x * o4.x + p0.y * o4.y + p0.z * o4.z + p0.w * o4.w;
                l1 += p1.x * o4.x + p1.y * o4.y + p1.z * o4.z + p1.w * o4.w;
            }
        }
        // softmax over I=512 (block-wide)
        float mx = fmaxf(l0, l1);
        #pragma unroll
        for (int m = 16; m; m >>= 1) mx = fmaxf(mx, __shfl_xor(mx, m, 32));
        if (lane == 0) wred[wv] = mx;
        __syncthreads();
        mx = wred[0];
        #pragma unroll
        for (int k = 1; k < 8; ++k) mx = fmaxf(mx, wred[k]);
        float e0 = __expf(l0 - mx), e1 = __expf(l1 - mx);
        float sm = e0 + e1;
        #pragma unroll
        for (int m = 16; m; m >>= 1) sm += __shfl_xor(sm, m, 32);
        __syncthreads();
        if (lane == 0) wred[wv] = sm;
        __syncthreads();
        sm = wred[0] + wred[1] + wred[2] + wred[3] + wred[4] + wred[5] + wred[6] + wred[7];
        float inv = 1.f / sm;
        coef[tid]       = e0 * inv;
        coef[tid + 256] = e1 * inv;
        __syncthreads();

        // out_j = sum_i coef_i * priors[i,j]  (float4 column partial sums)
        {
            float4 s = {0.f, 0.f, 0.f, 0.f};
            for (int i = qq * 32; i < qq * 32 + 32; ++i) {
                float  c = coef[i];
                float4 p = *(const float4*)(priors + i * STRIDE + jg * 4);
                s.x += c * p.x; s.y += c * p.y; s.z += c * p.z; s.w += c * p.w;
            }
            *(float4*)(red + qq * 64 + jg * 4) = s;
        }
        __syncthreads();
        if (tid < 64) {
            float s = 0.f;
            #pragma unroll
            for (int q2 = 0; q2 < 16; ++q2) s += red[q2 * 64 + tid];
            outv[tid] = s;
        }
        __syncthreads();
    }

    if (tid < 64)
        out[((size_t)b * O_ + o) * J_ + tid] = outv[tid];
}

extern "C" void kernel_launch(void* const* d_in, const int* in_sizes, int n_in,
                              void* d_out, int out_size, void* d_ws, size_t ws_size,
                              hipStream_t stream) {
    (void)in_sizes; (void)n_in; (void)out_size;
    const float* x = (const float*)d_in[0];   // (32, 512, 64)
    const float* w = (const float*)d_in[1];   // (128, 64, 64)
    float* out = (float*)d_out;               // (32, 128, 64)

    const size_t NX = (size_t)B_ * I_ * L_;   // 1,048,576
    const size_t NW = (size_t)O_ * J_ * L_;   //   524,288
    const size_t need = (NX + NW) * sizeof(unsigned short);   // 3 MB

    const size_t shmem = (size_t)LDS_FLOATS * sizeof(float);  // ~145.9 KB (2 WGs / 320 KB WGP)
    static_assert(2 * LDS_FLOATS * sizeof(float) < 320 * 1024, "LDS budget");

    dim3 grid(O_, B_);  // 128 x 32 = 4096 workgroups

    if (ws_size >= need) {
        unsigned short* xb16 = (unsigned short*)d_ws;
        unsigned short* wb16 = xb16 + NX;
        cvt_bf16_kernel<<<(int)(NX / 4 / 256), 256, 0, stream>>>(x, xb16, (int)(NX / 4));
        cvt_bf16_kernel<<<(int)(NW / 4 / 256), 256, 0, stream>>>(w, wb16, (int)(NW / 4));
        hipFuncSetAttribute((const void*)capsule_kmeans_kernel<true>,
                            hipFuncAttributeMaxDynamicSharedMemorySize, (int)shmem);
        capsule_kmeans_kernel<true><<<grid, 256, shmem, stream>>>(x, w, xb16, wb16, out);
    } else {
        hipFuncSetAttribute((const void*)capsule_kmeans_kernel<false>,
                            hipFuncAttributeMaxDynamicSharedMemorySize, (int)shmem);
        capsule_kmeans_kernel<false><<<grid, 256, shmem, stream>>>(x, w, nullptr, nullptr, out);
    }
}
